// _UNetLayer_long_89653147336725
// MI455X (gfx1250) — matrix operand order repaired and checked by reading, not compiled, not run on hardware
//
#include <hip/hip_runtime.h>

typedef _Float16 half_t;
typedef __attribute__((ext_vector_type(16))) _Float16 v16h;
typedef __attribute__((ext_vector_type(8)))  _Float16 v8h;
typedef __attribute__((ext_vector_type(8)))  float    v8f;

#define CIN   32
#define COUT  64
#define TT    2
#define S     48            // D = H = W
#define SP    50            // padded spatial
#define TP    4             // padded T
#define NSPAT (S*S*S)       // 110592
#define KOFFS 81            // 3*3*3*3 taps
#define NH    4             // output h-rows per wave
#define TROWS 6             // LDS tile rows   (kh + j : 0..5)
#define TPOS  18            // LDS tile w-pos  (n + kw : 0..17)

// workspace byte offsets
#define WS_STATS 0          // 64*2 floats = 512 B
#define WS_W     1024       // 81*64*32 halves = 331776 B
#define WS_ACT   332800     // 4*50*50*50*32 halves = 32,000,000 B

// ---------------- Stage 1: per-(c,t) mean / rstd ----------------
__global__ void stats_kernel(const float* __restrict__ x, float* __restrict__ stats) {
    __shared__ float s_sum[256];
    __shared__ float s_sq[256];
    int g = blockIdx.x;                       // g = c*T + t
    const float* p = x + (size_t)g * NSPAT;
    float sum = 0.f, sq = 0.f;
    for (int i = threadIdx.x; i < NSPAT; i += 256) {
        float v = p[i];
        sum += v; sq += v * v;
    }
    s_sum[threadIdx.x] = sum;
    s_sq[threadIdx.x]  = sq;
    __syncthreads();
    for (int off = 128; off > 0; off >>= 1) {
        if ((int)threadIdx.x < off) {
            s_sum[threadIdx.x] += s_sum[threadIdx.x + off];
            s_sq[threadIdx.x]  += s_sq[threadIdx.x + off];
        }
        __syncthreads();
    }
    if (threadIdx.x == 0) {
        float mean = s_sum[0] / (float)NSPAT;
        float var  = s_sq[0] / (float)NSPAT - mean * mean;   // biased, torch-style
        stats[2*g]   = mean;
        stats[2*g+1] = rsqrtf(var + 1e-5f);
    }
}

// ---------------- Stage 2: weights fp32 -> f16 [tap][cout][cin] ----------------
__global__ void pack_w_kernel(const float* __restrict__ Wk, half_t* __restrict__ wf) {
    int i = blockIdx.x * 256 + threadIdx.x;   // over COUT*CIN*KOFFS
    if (i >= COUT * CIN * KOFFS) return;
    int koff = i % KOFFS;                     // src layout (cout,cin,kt,kd,kh,kw) -> tap fastest
    int cin  = (i / KOFFS) % CIN;
    int cout = i / (KOFFS * CIN);
    wf[((size_t)koff * COUT + cout) * CIN + cin] = (half_t)Wk[i];
}

// ---------------- Stage 3: norm + ReLU + f16, zero-padded channel-last ----------------
__global__ void pack_act_kernel(const float* __restrict__ x, const float* __restrict__ stats,
                                unsigned* __restrict__ act) {
    const int TOT = TP * SP * SP * SP * (CIN / 2);
    int i = blockIdx.x * 256 + threadIdx.x;
    if (i >= TOT) return;
    int c2 = i & 15;
    int r  = i >> 4;
    int wp = r % SP; r /= SP;
    int hp = r % SP; r /= SP;
    int dp = r % SP;
    int tp = r / SP;
    unsigned outv = 0u;
    int t = tp - 1, d = dp - 1, h = hp - 1, w = wp - 1;
    if (t >= 0 && t < TT && d >= 0 && d < S && h >= 0 && h < S && w >= 0 && w < S) {
        union { half_t h2[2]; unsigned u; } pk;
        #pragma unroll
        for (int j = 0; j < 2; ++j) {
            int c = c2 * 2 + j;
            int g = c * TT + t;
            float v = x[(size_t)g * NSPAT + ((size_t)d * S + h) * S + w];
            v = (v - stats[2*g]) * stats[2*g + 1];
            v = v > 0.f ? v : 0.f;                         // ReLU
            pk.h2[j] = (half_t)v;
        }
        outv = pk.u;
    }
    act[i] = outv;
}

// ---------------- Stage 4: implicit-GEMM conv via WMMA + async LDS pipeline ----
// Per WG: 64 couts (4 waves x 16) x 16 w-positions x NH=4 h-rows.
// 9 stages (kt,kd); each stage's activation tile (6 rows x 18 pos x 32 cin, 6.9 KB)
// is async-copied to LDS one stage ahead (double buffered). Per stage, 18 unique
// B fragments are ds-loaded and reused across (kh,j) pairs -> 36 WMMAs/stage.
__global__ void __launch_bounds__(128)
conv_wmma_kernel(const half_t* __restrict__ wf, const half_t* __restrict__ act,
                 const float* __restrict__ bias, float* __restrict__ out) {
    __shared__ __attribute__((aligned(64))) half_t sbuf[2][TROWS][TPOS][CIN];

    int tid  = threadIdx.x;
    int lane = tid & 31;
    int wave = tid >> 5;
    int coutBase = wave * 16;

    int w0 = blockIdx.x * 16;
    int h0 = blockIdx.y * NH;
    int d  = blockIdx.z >> 1;
    int t  = blockIdx.z & 1;

    int n      = lane & 15;        // B/C/D column (output w); also A row M
    int hiHalf = lane >> 4;        // lane-group select
    int cinOff = hiHalf * 16;      // B-fragment K base: {0,16}
    int k0     = hiHalf * 16;      // A-fragment K base: {0,16} (contiguous 16)

    v8f acc[NH];
    #pragma unroll
    for (int j = 0; j < NH; ++j) acc[j] = (v8f){};

    const half_t* wp0 = wf + ((size_t)(coutBase + n) * CIN + k0);

    // Issue the async copy of stage s's activation tile into sbuf[s&1].
    // Tile item i (16 B) : row = i/72, pos = (i%72)>>2, 16B-chunk = i&3.
    auto issue_copy = [&](int s) {
        int kt = s / 3, kd = s % 3;
        const half_t* gb = act
            + ((((size_t)(t + kt) * SP + (d + kd)) * SP + h0) * SP + w0) * CIN;
        unsigned lbase = (unsigned)(uintptr_t)&sbuf[s & 1][0][0][0];
        for (int i = tid; i < TROWS * TPOS * 4; i += 128) {
            int row = i / (TPOS * 4);
            int rem = i - row * (TPOS * 4);
            int pos = rem >> 2;
            int cp  = rem & 3;
            const half_t* g = gb + ((size_t)row * SP + pos) * CIN + cp * 8;
            unsigned l = lbase + (unsigned)i * 16;
            asm volatile("global_load_async_to_lds_b128 %0, %1, off"
                         :: "v"(l), "v"(g) : "memory");
        }
    };

    issue_copy(0);
    asm volatile("s_wait_asynccnt 0x0" ::: "memory");
    __syncthreads();

    for (int s = 0; s < 9; ++s) {
        if (s + 1 < 9) {
            issue_copy(s + 1);                                   // overlap with compute
            __builtin_prefetch(wp0 + (size_t)(9 * (s + 1)) * (COUT * CIN), 0, 1);
        }
        const half_t* lb = &sbuf[s & 1][0][0][0];

        #pragma unroll
        for (int kw = 0; kw < 3; ++kw) {
            // 6 unique B row-fragments for this kw (row r used by all kh+j==r)
            v16h bf[TROWS];
            #pragma unroll
            for (int r6 = 0; r6 < TROWS; ++r6)
                bf[r6] = *(const v16h*)(lb + ((size_t)(r6 * TPOS) + (n + kw)) * CIN
                                            + cinOff);
            #pragma unroll
            for (int kh = 0; kh < 3; ++kh) {
                const half_t* pa = wp0
                    + (size_t)(9 * s + 3 * kh + kw) * (COUT * CIN);
                v8h alo = *(const v8h*)(pa);
                v8h ahi = *(const v8h*)(pa + 8);
                v16h a = __builtin_shufflevector(alo, ahi,
                             0,1,2,3,4,5,6,7,8,9,10,11,12,13,14,15);
                #pragma unroll
                for (int j = 0; j < NH; ++j)
                    acc[j] = __builtin_amdgcn_wmma_f32_16x16x32_f16(
                                 false, a, false, bf[kh + j], (short)0, acc[j],
                                 false, false);
            }
        }
        asm volatile("s_wait_asynccnt 0x0" ::: "memory");   // next tile landed
        __syncthreads();                                    // all waves done w/ buffers
    }

    // C/D layout: VGPR r -> M = r + 8*hiHalf, N = n. Non-temporal write-once stream.
    #pragma unroll
    for (int j = 0; j < NH; ++j) {
        int h = h0 + j;
        #pragma unroll
        for (int r = 0; r < 8; ++r) {
            int cout = coutBase + r + hiHalf * 8;
            size_t oidx = (size_t)(cout * TT + t) * NSPAT
                        + ((size_t)d * S + h) * S + (w0 + n);
            __builtin_nontemporal_store(acc[j][r] + bias[cout], &out[oidx]);
        }
    }
}

extern "C" void kernel_launch(void* const* d_in, const int* in_sizes, int n_in,
                              void* d_out, int out_size, void* d_ws, size_t ws_size,
                              hipStream_t stream) {
    const float* x  = (const float*)d_in[0];
    const float* Wk = (const float*)d_in[1];
    const float* b  = (const float*)d_in[2];
    float* out = (float*)d_out;

    char* ws = (char*)d_ws;
    float*  stats = (float*)(ws + WS_STATS);
    half_t* wf    = (half_t*)(ws + WS_W);
    half_t* act   = (half_t*)(ws + WS_ACT);

    stats_kernel<<<64, 256, 0, stream>>>(x, stats);

    int nw = COUT * CIN * KOFFS;
    pack_w_kernel<<<(nw + 255) / 256, 256, 0, stream>>>(Wk, wf);

    int na = TP * SP * SP * SP * (CIN / 2);
    pack_act_kernel<<<(na + 255) / 256, 256, 0, stream>>>(x, stats, (unsigned*)act);

    dim3 grid(S / 16, S / NH, S * TT);   // (3, 12, 96)
    conv_wmma_kernel<<<grid, 128, 0, stream>>>(wf, act, b, out);
}
